// Model_16140487099104
// MI455X (gfx1250) — matrix operand order repaired and checked by reading, not compiled, not run on hardware
//
#include <hip/hip_runtime.h>
#include <hip/hip_bf16.h>

typedef unsigned int  u32;
typedef unsigned short u16;

typedef __bf16 v16bf __attribute__((ext_vector_type(16)));
typedef float  v8f   __attribute__((ext_vector_type(8)));

union Frag { v16bf v; u32 u[8]; };

#define NUM_ENT   50000
#define DIM       2000
#define BATCH     1024
#define KPAD      2048          // x1 workspace K stride (bf16 elems)
#define KLOOP     2016          // 63 * 32 >= 2000
#define BM        64
#define BN        80
#define BK        32
#define LDT       40            // LDS row stride in ushorts (80B, conflict-friendly)

__device__ __forceinline__ u16 f2bf(float f) {
    u32 u = __float_as_uint(f);
    u32 r = u + 0x7FFFu + ((u >> 16) & 1u);    // round-to-nearest-even
    return (u16)(r >> 16);
}
__device__ __forceinline__ float bf2f(u16 h) {
    return __uint_as_float(((u32)h) << 16);
}

// ---------------------------------------------------------------------------
// Kernel 1: gather h/r/t, build x1 = h (complex*) r as bf16 hi/lo (K-padded),
// and per-sample factor partials s1..s4.  One block per sample.
// ---------------------------------------------------------------------------
__global__ __launch_bounds__(256)
void build_x1_norms(const int* __restrict__ heads, const int* __restrict__ rels,
                    const int* __restrict__ tails,
                    const float* __restrict__ E, const float* __restrict__ R,
                    u32* __restrict__ x1h, u32* __restrict__ x1l,
                    float* __restrict__ partials)
{
    const int b   = blockIdx.x;
    const int tid = threadIdx.x;

    const float2* hrow = (const float2*)(E + (size_t)heads[b] * DIM);
    const float2* rrow = (const float2*)(R + (size_t)rels[b]  * DIM);
    const float2* trow = (const float2*)(E + (size_t)tails[b] * DIM);

    float s1 = 0.f, s2 = 0.f, s3 = 0.f, s4 = 0.f;

    for (int d = tid; d < KPAD / 2; d += 256) {
        u32 uh = 0u, ul = 0u;
        if (d < DIM / 2) {
            float2 h = hrow[d], r = rrow[d], t = trow[d];
            float h2 = h.x * h.x + h.y * h.y;
            float r2 = r.x * r.x + r.y * r.y;
            float t2 = t.x * t.x + t.y * t.y;
            // x1 = h * r   (ComplEx W contraction)
            float x10 = h.x * r.x - h.y * r.y;
            float x11 = h.x * r.y + h.y * r.x;
            // x2 = conj(r) * t
            float x20 = t.x * r.x + t.y * r.y;
            float x21 = t.y * r.x - t.x * r.y;
            // x3 = conj(h)-style contraction with t
            float x30 = h.x * t.x + h.y * t.y;
            float x31 = h.x * t.y - h.y * t.x;

            s1 += h2 + r2 + t2;
            s2 += h2 * r2 + r2 * t2 + t2 * h2;
            s3 += h2 + r2 + t2;                    // w{h,r,t}_norm reduce to same sums
            s4 += (x10 * x10 + x11 * x11) + (x20 * x20 + x21 * x21)
                + (x30 * x30 + x31 * x31);

            u16 b0 = f2bf(x10); u16 b1 = f2bf(x11);
            u16 l0 = f2bf(x10 - bf2f(b0));
            u16 l1 = f2bf(x11 - bf2f(b1));
            uh = (u32)b0 | ((u32)b1 << 16);
            ul = (u32)l0 | ((u32)l1 << 16);
        }
        x1h[(size_t)b * (KPAD / 2) + d] = uh;   // zero pad K in [2000, 2048)
        x1l[(size_t)b * (KPAD / 2) + d] = ul;
    }

    __shared__ float red[256 * 4];
    red[tid * 4 + 0] = s1; red[tid * 4 + 1] = s2;
    red[tid * 4 + 2] = s3; red[tid * 4 + 3] = s4;
    __syncthreads();
    for (int s = 128; s > 0; s >>= 1) {
        if (tid < s) {
            #pragma unroll
            for (int i = 0; i < 4; ++i)
                red[tid * 4 + i] += red[(tid + s) * 4 + i];
        }
        __syncthreads();
    }
    if (tid == 0) {
        #pragma unroll
        for (int i = 0; i < 4; ++i) partials[b * 4 + i] = red[i];
    }
}

// ---------------------------------------------------------------------------
// Kernel 2: scores = x1(1024 x 2000) @ E^T(2000 x 50000) via bf16 WMMA with
// hi/lo split (3 x v_wmma_f32_16x16x32_bf16 per 16x16xK32 tile).
// Block tile 64x80, 128 threads = 4 wave32; wave w owns M-subtile w and five
// N-subtiles.  Entity tiles converted fp32 -> bf16 hi/lo in LDS.
// ---------------------------------------------------------------------------
__global__ __launch_bounds__(128)
void gemm_scores(const u32* __restrict__ x1h, const u32* __restrict__ x1l,
                 const float* __restrict__ E, float* __restrict__ scores)
{
    __shared__ __align__(16) u16 sAh[BM * LDT];
    __shared__ __align__(16) u16 sAl[BM * LDT];
    __shared__ __align__(16) u16 sBh[BN * LDT];
    __shared__ __align__(16) u16 sBl[BN * LDT];

    const int tid  = threadIdx.x;
    const int wave = tid >> 5;          // wave32
    const int lane = tid & 31;
    const int m0   = blockIdx.x * BM;   // 16 M-blocks (fast dim -> L2 reuse of E)
    const int n0   = blockIdx.y * BN;   // 625 N-blocks (50000 / 80 exact)

    v8f acc[5];
    #pragma unroll
    for (int j = 0; j < 5; ++j)
        #pragma unroll
        for (int i = 0; i < 8; ++i) acc[j][i] = 0.f;

    // Per-lane WMMA fragment addressing (ISA 7.12.2, wave32)
    const int am   = lane & 15;               // A row within subtile
    const int akh  = (lane >> 4) * 16;        // A K-half select
    const int bn   = lane & 15;               // B column within subtile
    const int bkof = (lane >> 4) * 16;        // B K-offset

    for (int kk = 0; kk < KLOOP; kk += BK) {
        // ---- stage A tile (already bf16 hi/lo in workspace) ----
        {
            const int row = tid >> 1;                  // 0..63
            const int hsel = tid & 1;                  // which 16-elem K half
            const u32* srch = x1h + (size_t)(m0 + row) * (KPAD / 2) + (kk >> 1) + hsel * 8;
            const u32* srcl = x1l + (size_t)(m0 + row) * (KPAD / 2) + (kk >> 1) + hsel * 8;
            u32* dh = (u32*)sAh + row * (LDT / 2) + hsel * 8;
            u32* dl = (u32*)sAl + row * (LDT / 2) + hsel * 8;
            #pragma unroll
            for (int i = 0; i < 8; ++i) { dh[i] = srch[i]; dl[i] = srcl[i]; }
        }
        // ---- stage B tile: fp32 entity rows -> bf16 hi/lo ----
        #pragma unroll
        for (int pass = 0; pass < 5; ++pass) {
            const int nl = pass * 16 + (tid >> 3);     // 0..79
            const int kb = (tid & 7) * 4;              // 0,4,...,28
            const int gk = kk + kb;
            const float* src = E + (size_t)(n0 + nl) * DIM + gk;
            float v[4];
            #pragma unroll
            for (int j = 0; j < 4; ++j)
                v[j] = (gk + j < DIM) ? src[j] : 0.f;  // K pad guard
            u32* dh = (u32*)sBh + nl * (LDT / 2) + (kb >> 1);
            u32* dl = (u32*)sBl + nl * (LDT / 2) + (kb >> 1);
            #pragma unroll
            for (int j = 0; j < 2; ++j) {
                u16 h0 = f2bf(v[2 * j]),     h1 = f2bf(v[2 * j + 1]);
                u16 l0 = f2bf(v[2 * j]     - bf2f(h0));
                u16 l1 = f2bf(v[2 * j + 1] - bf2f(h1));
                dh[j] = (u32)h0 | ((u32)h1 << 16);
                dl[j] = (u32)l0 | ((u32)l1 << 16);
            }
        }
        __syncthreads();

        // ---- A fragments for this wave's M-subtile ----
        Frag ah, al;
        {
            const u32* Ah = (const u32*)sAh;
            const u32* Al = (const u32*)sAl;
            const int r  = wave * 16 + am;
            const int b0i = (r * LDT + akh) >> 1;          // K = akh .. akh+7
            const int b1i = (r * LDT + 8 + akh) >> 1;      // K = 8+akh .. 8+akh+7
            #pragma unroll
            for (int i = 0; i < 4; ++i) {
                ah.u[i]     = Ah[b0i + i];  al.u[i]     = Al[b0i + i];
                ah.u[4 + i] = Ah[b1i + i];  al.u[4 + i] = Al[b1i + i];
            }
        }
        // ---- 5 N-subtiles, 3 WMMAs each (hi*hi + hi*lo + lo*hi) ----
        const u32* Bh = (const u32*)sBh;
        const u32* Bl = (const u32*)sBl;
        #pragma unroll
        for (int j = 0; j < 5; ++j) {
            Frag bh, bl;
            const int bi = ((j * 16 + bn) * LDT + bkof) >> 1;
            #pragma unroll
            for (int i = 0; i < 8; ++i) { bh.u[i] = Bh[bi + i]; bl.u[i] = Bl[bi + i]; }
            acc[j] = __builtin_amdgcn_wmma_f32_16x16x32_bf16(
                         false, ah.v, false, bh.v, (short)0, acc[j], false, false);
            acc[j] = __builtin_amdgcn_wmma_f32_16x16x32_bf16(
                         false, ah.v, false, bl.v, (short)0, acc[j], false, false);
            acc[j] = __builtin_amdgcn_wmma_f32_16x16x32_bf16(
                         false, al.v, false, bh.v, (short)0, acc[j], false, false);
        }
        __syncthreads();
    }

    // ---- epilogue: C/D layout -> global scores ----
    const int mrow0 = m0 + wave * 16 + (lane >> 4) * 8;
    const int ncol  = lane & 15;
    #pragma unroll
    for (int j = 0; j < 5; ++j) {
        const size_t col = (size_t)(n0 + j * 16 + ncol);
        #pragma unroll
        for (int r = 0; r < 8; ++r)
            scores[(size_t)(mrow0 + r) * NUM_ENT + col] = acc[j][r];
    }
}

// ---------------------------------------------------------------------------
// Kernel 3: deterministic reduction of per-sample partials -> factor1..4
// ---------------------------------------------------------------------------
__global__ __launch_bounds__(256)
void reduce_factors(const float* __restrict__ partials, float* __restrict__ out)
{
    __shared__ float red[256 * 4];
    const int tid = threadIdx.x;
    float a[4] = {0.f, 0.f, 0.f, 0.f};
    for (int b = tid; b < BATCH; b += 256) {
        #pragma unroll
        for (int i = 0; i < 4; ++i) a[i] += partials[b * 4 + i];
    }
    #pragma unroll
    for (int i = 0; i < 4; ++i) red[tid * 4 + i] = a[i];
    __syncthreads();
    for (int s = 128; s > 0; s >>= 1) {
        if (tid < s) {
            #pragma unroll
            for (int i = 0; i < 4; ++i)
                red[tid * 4 + i] += red[(tid + s) * 4 + i];
        }
        __syncthreads();
    }
    if (tid == 0) {
        #pragma unroll
        for (int i = 0; i < 4; ++i) out[i] = red[i] * (1.0f / (float)BATCH);
    }
}

// ---------------------------------------------------------------------------
extern "C" void kernel_launch(void* const* d_in, const int* in_sizes, int n_in,
                              void* d_out, int out_size, void* d_ws, size_t ws_size,
                              hipStream_t stream)
{
    const int*   heads = (const int*)d_in[0];
    const int*   rels  = (const int*)d_in[1];
    const int*   tails = (const int*)d_in[2];
    const float* E     = (const float*)d_in[3];
    const float* R     = (const float*)d_in[4];
    float* out = (float*)d_out;

    char* ws = (char*)d_ws;
    u32*   x1h      = (u32*)ws;                                   // 4 MB
    u32*   x1l      = (u32*)(ws + (size_t)BATCH * (KPAD / 2) * 4);// 4 MB
    float* partials = (float*)(ws + (size_t)2 * BATCH * (KPAD / 2) * 4); // 16 KB

    build_x1_norms<<<BATCH, 256, 0, stream>>>(heads, rels, tails, E, R,
                                              x1h, x1l, partials);

    dim3 grid(BATCH / BM, NUM_ENT / BN);   // (16, 625): x fastest -> E-tile L2 reuse
    gemm_scores<<<grid, 128, 0, stream>>>(x1h, x1l, E, out);

    reduce_factors<<<1, 256, 0, stream>>>(partials, out + (size_t)BATCH * NUM_ENT);
}
